// RNNTDecodingWrapper_6614249636149
// MI455X (gfx1250) — compile-verified, matrix operand order checked by reading, not run on hardware
//
#include <hip/hip_runtime.h>
#include <hip/hip_bf16.h>
#include <math.h>

// ---------------------------------------------------------------------------
// RNN-T decode on CDNA5 (gfx1250, wave32):
//   - cooperative grid-synced LSTM (packed f16 VALU, 40 WGs)
//   - WMMA f16 projections f_enc / f_pred
//   - fused relu(f_enc+f_pred) @ out_w joint GEMM in v_wmma_f32_16x16x32_f16
// ---------------------------------------------------------------------------

typedef _Float16 v16h __attribute__((ext_vector_type(16)));
typedef _Float16 v8h  __attribute__((ext_vector_type(8)));
typedef float    v8f  __attribute__((ext_vector_type(8)));

#define B_   4
#define T_   256
#define U1   65          // U+1
#define H_   640         // PRED_H == JOINT_D
#define ED_  512         // ENC_D
#define VO_  1025        // V+1
#define LG_  40          // LSTM cooperative grid (co-resident; 10240 threads)

// Load one WMMA f16 operand fragment for 16x16x32: per lane, two contiguous
// 16-byte chunks of a row-major row at k0 and k0+16 halves.
__device__ __forceinline__ v16h ld_frag(const _Float16* p) {
  v8h lo = *(const v8h*)(p);
  v8h hi = *(const v8h*)(p + 16);
  return __builtin_shufflevector(lo, hi, 0,1,2,3,4,5,6,7,8,9,10,11,12,13,14,15);
}

__device__ __forceinline__ v8h relu8(v8h x) {
  v8h z = {0,0,0,0,0,0,0,0};
  return __builtin_elementwise_max(x, z);
}

// --------------------------- conversion kernels ----------------------------

__global__ void f32_to_f16_kernel(const float* __restrict__ src,
                                  _Float16* __restrict__ dst, int n) {
  int i = blockIdx.x * blockDim.x + threadIdx.x;
  if (i < n) dst[i] = (_Float16)src[i];
}

// Combined LSTM weights: wC[row][0..639] = w_ih[row], wC[row][640..1279] = w_hh[row]
__global__ void build_wc_kernel(const float* __restrict__ w_ih,
                                const float* __restrict__ w_hh,
                                _Float16* __restrict__ wC) {
  int i = blockIdx.x * blockDim.x + threadIdx.x;   // 2560*1280
  if (i >= 4 * H_ * 2 * H_) return;
  int row = i / (2 * H_), k = i % (2 * H_);
  float v = (k < H_) ? w_ih[(size_t)row * H_ + k] : w_hh[(size_t)row * H_ + (k - H_)];
  wC[i] = (_Float16)v;
}

// encoder_output [B, ED, T] f32  ->  encT [B*T, ED] f16
__global__ void enc_transpose_kernel(const float* __restrict__ enc,
                                     _Float16* __restrict__ encT) {
  int bk = blockIdx.x;            // b*ED + k
  int b = bk / ED_, k = bk % ED_;
  int t = threadIdx.x;            // 0..255 (coalesced read)
  float v = enc[((size_t)b * ED_ + k) * T_ + t];
  encT[((size_t)b * T_ + t) * ED_ + k] = (_Float16)v;
}

// --------------------- cooperative grid-wide LSTM --------------------------

__device__ __forceinline__ void grid_barrier(int* cnt, int* gen, int nblk) {
  __threadfence();                         // flush this thread's stores to L2
  __syncthreads();
  if (threadIdx.x == 0) {
    int g = __hip_atomic_load(gen, __ATOMIC_RELAXED, __HIP_MEMORY_SCOPE_AGENT);
    int arrived = __hip_atomic_fetch_add(cnt, 1, __ATOMIC_ACQ_REL,
                                         __HIP_MEMORY_SCOPE_AGENT);
    if (arrived == nblk - 1) {
      __hip_atomic_store(cnt, 0, __ATOMIC_RELAXED, __HIP_MEMORY_SCOPE_AGENT);
      __hip_atomic_fetch_add(gen, 1, __ATOMIC_RELEASE, __HIP_MEMORY_SCOPE_AGENT);
    } else {
      while (__hip_atomic_load(gen, __ATOMIC_ACQUIRE,
                               __HIP_MEMORY_SCOPE_AGENT) == g) {
        __builtin_amdgcn_s_sleep(1);
      }
    }
  }
  __syncthreads();
}

// One thread per gate value: tid -> (b, unit u, gate g); the 4 gates of a unit
// sit on adjacent lanes of a wave32 and are combined with __shfl.
// h_t is written straight into pred16 (the joint stage's f_pred input).
__global__ __launch_bounds__(256) void lstm_coop_kernel(
    const int*      __restrict__ targets,  // [B, 64]
    const _Float16* __restrict__ embed16,  // [V+1, H]
    const _Float16* __restrict__ wC16,     // [4H, 2H] f16  (w_ih | w_hh)
    const float*    __restrict__ b_ih,     // [4H]
    const float*    __restrict__ b_hh,     // [4H]
    _Float16*       __restrict__ pred16,   // [B, U1, H]
    int*            __restrict__ bar) {    // bar[0]=cnt, bar[1]=gen (zeroed)
  const int tid = blockIdx.x * 256 + threadIdx.x;   // 0..10239
  const int b = tid / (4 * H_);
  const int r = tid % (4 * H_);
  const int u = r >> 2;
  const int g = r & 3;
  const int row = g * H_ + u;                       // torch gate order i,f,g,o
  const int lane = threadIdx.x & 31;

  const float bias = b_ih[row] + b_hh[row];
  const v8h* wx = (const v8h*)(wC16 + (size_t)row * (2 * H_));  // 80 v8h
  const v8h* wh = wx + (H_ / 8);

  float c = 0.f;
  for (int t = 0; t < U1; ++t) {
    float gv = bias;
    if (t > 0) {
      int tok = targets[b * 64 + (t - 1)];
      const v8h* xr = (const v8h*)(embed16 + (size_t)tok * H_);
      const v8h* hr = (const v8h*)(pred16 + ((size_t)b * U1 + (t - 1)) * H_);
      v8h acc = {0,0,0,0,0,0,0,0};
      #pragma unroll 8
      for (int k = 0; k < H_ / 8; ++k)
        acc = __builtin_elementwise_fma(xr[k], wx[k], acc);   // v_pk_fma_f16
      #pragma unroll 8
      for (int k = 0; k < H_ / 8; ++k)
        acc = __builtin_elementwise_fma(hr[k], wh[k], acc);
      float s = 0.f;
      #pragma unroll
      for (int i = 0; i < 8; ++i) s += (float)acc[i];
      gv += s;
    }
    // gather the quartet's gate values (i,f,g,o)
    const int base = lane & ~3;
    float gi = __shfl(gv, base + 0, 32);
    float gf = __shfl(gv, base + 1, 32);
    float gg = __shfl(gv, base + 2, 32);
    float go = __shfl(gv, base + 3, 32);
    float si = 1.f / (1.f + __expf(-gi));
    float sf = 1.f / (1.f + __expf(-gf));
    float so = 1.f / (1.f + __expf(-go));
    c = sf * c + si * tanhf(gg);        // all 4 lanes keep identical c
    float hn = so * tanhf(c);
    if (g == 0) pred16[((size_t)b * U1 + t) * H_ + u] = (_Float16)hn;
    if (t < U1 - 1) grid_barrier(bar, bar + 1, LG_);
  }
}

// ---------------------- generic small WMMA GEMM ----------------------------
// C[m,n] = sum_k A[m,k]*W[n,k] + bias[n], f16 in, f16 out.
// grid = (ceil(M/16), N/64), block = 32 (one wave -> 16x64 tile).
__global__ __launch_bounds__(32) void gemm16_kernel(
    const _Float16* __restrict__ A, const _Float16* __restrict__ W,
    const float* __restrict__ bias, _Float16* __restrict__ C,
    int M, int N, int K) {
  const int lane = threadIdx.x & 31;
  const int l16 = lane & 15, lh = lane >> 4;
  const int m = blockIdx.x * 16 + l16;
  const int mc = m < M ? m : M - 1;           // clamp: EXEC stays full
  const int n0 = blockIdx.y * 64;

  const _Float16* a = A + (size_t)mc * K;
  const _Float16* w0 = W + (size_t)(n0 + 0 * 16 + l16) * K;
  const _Float16* w1 = W + (size_t)(n0 + 1 * 16 + l16) * K;
  const _Float16* w2 = W + (size_t)(n0 + 2 * 16 + l16) * K;
  const _Float16* w3 = W + (size_t)(n0 + 3 * 16 + l16) * K;

  v8f acc0 = {}, acc1 = {}, acc2 = {}, acc3 = {};
  for (int kk = 0; kk < K; kk += 32) {
    int kb = kk + lh * 8;
    v16h af = ld_frag(a + kb);
    acc0 = __builtin_amdgcn_wmma_f32_16x16x32_f16(false, af, false, ld_frag(w0 + kb), (short)0, acc0, false, false);
    acc1 = __builtin_amdgcn_wmma_f32_16x16x32_f16(false, af, false, ld_frag(w1 + kb), (short)0, acc1, false, false);
    acc2 = __builtin_amdgcn_wmma_f32_16x16x32_f16(false, af, false, ld_frag(w2 + kb), (short)0, acc2, false, false);
    acc3 = __builtin_amdgcn_wmma_f32_16x16x32_f16(false, af, false, ld_frag(w3 + kb), (short)0, acc3, false, false);
  }

  v8f accs[4] = {acc0, acc1, acc2, acc3};
  #pragma unroll
  for (int j = 0; j < 4; ++j) {
    int n = n0 + j * 16 + l16;
    float bv = bias[n];
    #pragma unroll
    for (int r = 0; r < 8; ++r) {
      int row = blockIdx.x * 16 + r + 8 * lh;
      if (row < M) C[(size_t)row * N + n] = (_Float16)(accs[j][r] + bv);
    }
  }
}

// -------------------------- fused joint kernel -----------------------------
// logits[b,t,u,n] = sum_k relu(f_enc[b,t,k]+f_pred[b,u,k]) * out_w[n,k] + out_b[n]
// grid = (130, 9, B), block = 256 = 8 waves as 4(M) x 2(N);
// each wave: 2 M-frags x 4 N-frags (A-operand pk-work halved vs 4x2).
__global__ __launch_bounds__(256) void joint_kernel(
    const _Float16* __restrict__ fenc,   // [B*T, H] f16 (incl. enc_b)
    const _Float16* __restrict__ fpred,  // [B*U1, H] f16 (incl. pred_b)
    const _Float16* __restrict__ wout,   // [VO, H] f16
    const float* __restrict__ outb,      // [VO]
    float* __restrict__ out) {           // [B, T, U1, VO] f32
  const int b    = blockIdx.z;
  const int lane = threadIdx.x & 31;
  const int wave = threadIdx.x >> 5;
  const int wn = wave & 1;               // 2 N-wave columns
  const int wm = wave >> 1;              // 4 M-wave rows
  const int l16 = lane & 15, lh = lane >> 4;
  const int mBase = blockIdx.x * 128 + wm * 32;   // 2 frags * 16
  const int nBase = blockIdx.y * 128 + wn * 64;   // 4 frags * 16

  const _Float16* aE[2];
  const _Float16* aP[2];
  #pragma unroll
  for (int i = 0; i < 2; ++i) {
    int m = mBase + i * 16 + l16;          // always < 16640
    int t = m / U1, u = m % U1;
    aE[i] = fenc  + ((size_t)b * T_ + t) * H_;
    aP[i] = fpred + ((size_t)b * U1 + u) * H_;
  }
  const _Float16* bW[4];
  float bb[4];
  bool  nOK[4];
  #pragma unroll
  for (int j = 0; j < 4; ++j) {
    int n = nBase + j * 16 + l16;
    int nc = n < VO_ ? n : VO_ - 1;        // clamp: EXEC stays full in loop
    bW[j] = wout + (size_t)nc * H_;
    bb[j] = outb[nc];
    nOK[j] = (n < VO_);
  }

  v8f acc[2][4] = {};
  for (int kk = 0; kk < H_; kk += 32) {
    const int kb = kk + lh * 8;
    v16h bf[4];
    #pragma unroll
    for (int j = 0; j < 4; ++j) bf[j] = ld_frag(bW[j] + kb);
    #pragma unroll
    for (int i = 0; i < 2; ++i) {
      v8h e0 = *(const v8h*)(aE[i] + kb);
      v8h e1 = *(const v8h*)(aE[i] + kb + 16);
      v8h p0 = *(const v8h*)(aP[i] + kb);
      v8h p1 = *(const v8h*)(aP[i] + kb + 16);
      v8h s0 = relu8(e0 + p0);             // v_pk_add_f16 + v_pk_max
      v8h s1 = relu8(e1 + p1);
      v16h af = __builtin_shufflevector(s0, s1, 0,1,2,3,4,5,6,7,8,9,10,11,12,13,14,15);
      #pragma unroll
      for (int j = 0; j < 4; ++j)
        acc[i][j] = __builtin_amdgcn_wmma_f32_16x16x32_f16(
            false, af, false, bf[j], (short)0, acc[i][j], false, false);
    }
  }

  // Epilogue: C VGPR r -> row (r + 8*(lane/16)), col = lane%16.
  #pragma unroll
  for (int i = 0; i < 2; ++i) {
    #pragma unroll
    for (int j = 0; j < 4; ++j) {
      if (!nOK[j]) continue;
      const int n = nBase + j * 16 + l16;
      const float bv = bb[j];
      #pragma unroll
      for (int r = 0; r < 8; ++r) {
        int m = mBase + i * 16 + r + 8 * lh;
        int t = m / U1, u = m % U1;
        size_t off = (((size_t)b * T_ + t) * U1 + u) * VO_ + n;
        out[off] = acc[i][j][r] + bv;
      }
    }
  }
}

// ------------------------------ launcher -----------------------------------

extern "C" void kernel_launch(void* const* d_in, const int* in_sizes, int n_in,
                              void* d_out, int out_size, void* d_ws, size_t ws_size,
                              hipStream_t stream) {
  const int*   targets = (const int*)  d_in[2];   // [B, 64]
  const float* embed   = (const float*)d_in[4];   // [VO, H]
  const float* w_ih    = (const float*)d_in[5];
  const float* w_hh    = (const float*)d_in[6];
  const float* b_ih    = (const float*)d_in[7];
  const float* b_hh    = (const float*)d_in[8];
  const float* enc_out = (const float*)d_in[0];   // [B, ED, T]
  const float* enc_w   = (const float*)d_in[9];   // [H, ED]
  const float* enc_b   = (const float*)d_in[10];
  const float* pred_w  = (const float*)d_in[11];  // [H, H]
  const float* pred_b  = (const float*)d_in[12];
  const float* out_w   = (const float*)d_in[13];  // [VO, H]
  const float* out_b   = (const float*)d_in[14];
  float* out = (float*)d_out;

  // Carve workspace (256B aligned).
  size_t off = 0;
  auto carve = [&](size_t bytes) {
    void* p = (char*)d_ws + off;
    off += (bytes + 255) & ~(size_t)255;
    return p;
  };
  _Float16* wEnc16  = (_Float16*)carve((size_t)H_ * ED_ * 2);
  _Float16* wPred16 = (_Float16*)carve((size_t)H_ * H_ * 2);
  _Float16* wOut16  = (_Float16*)carve((size_t)VO_ * H_ * 2);
  _Float16* encT16  = (_Float16*)carve((size_t)B_ * T_ * ED_ * 2);
  _Float16* pred16  = (_Float16*)carve((size_t)B_ * U1 * H_ * 2);
  _Float16* fenc16  = (_Float16*)carve((size_t)B_ * T_ * H_ * 2);
  _Float16* fpred16 = (_Float16*)carve((size_t)B_ * U1 * H_ * 2);
  _Float16* embed16 = (_Float16*)carve((size_t)VO_ * H_ * 2);
  _Float16* wC16    = (_Float16*)carve((size_t)4 * H_ * 2 * H_ * 2);
  int*      bar     = (int*)carve(256);
  (void)ws_size; (void)n_in; (void)in_sizes; (void)out_size;

  // 0) zero the grid-barrier state (graph-capture safe)
  hipMemsetAsync(bar, 0, 8, stream);

  // 1) f16 conversions
  {
    int n = H_ * ED_;
    f32_to_f16_kernel<<<(n + 255) / 256, 256, 0, stream>>>(enc_w, wEnc16, n);
    n = H_ * H_;
    f32_to_f16_kernel<<<(n + 255) / 256, 256, 0, stream>>>(pred_w, wPred16, n);
    n = VO_ * H_;
    f32_to_f16_kernel<<<(n + 255) / 256, 256, 0, stream>>>(out_w, wOut16, n);
    f32_to_f16_kernel<<<(n + 255) / 256, 256, 0, stream>>>(embed, embed16, n);
    n = 4 * H_ * 2 * H_;
    build_wc_kernel<<<(n + 255) / 256, 256, 0, stream>>>(w_ih, w_hh, wC16);
  }

  // 2) encoder transpose [B,ED,T] -> [B*T, ED] f16
  enc_transpose_kernel<<<B_ * ED_, T_, 0, stream>>>(enc_out, encT16);

  // 3) cooperative LSTM -> pred16 [B*U1, H] f16
  lstm_coop_kernel<<<LG_, 256, 0, stream>>>(targets, embed16, wC16,
                                            b_ih, b_hh, pred16, bar);

  // 4) f_enc = encT @ enc_w.T + enc_b : M=1024, N=640, K=512
  {
    dim3 g((B_ * T_) / 16, H_ / 64);
    gemm16_kernel<<<g, 32, 0, stream>>>(encT16, wEnc16, enc_b, fenc16,
                                        B_ * T_, H_, ED_);
  }
  // 5) f_pred = pred @ pred_w.T + pred_b : M=260, N=640, K=640
  {
    dim3 g((B_ * U1 + 15) / 16, H_ / 64);
    gemm16_kernel<<<g, 32, 0, stream>>>(pred16, wPred16, pred_b, fpred16,
                                        B_ * U1, H_, H_);
  }

  // 6) fused joint + output GEMM
  {
    dim3 g((T_ * U1) / 128, (VO_ + 127) / 128, B_);
    joint_kernel<<<g, 256, 0, stream>>>(fenc16, fpred16, wOut16, out_b, out);
  }
}
